// GnBlockONNX_53206054863198
// MI455X (gfx1250) — matrix operand order repaired and checked by reading, not compiled, not run on hardware
//
#include <hip/hip_runtime.h>

// MI455X (gfx1250) GraphNet block.
// bf16 WMMA (v_wmma_f32_16x16x32_bf16) with fp32 accumulation. One wave32
// computes a 16x128 tile per layer. Weight B-fragments are staged into LDS
// once per 4-wave block using the CDNA5 async global->LDS engine
// (global_load_async_to_lds_b128 / ASYNCcnt). Inter-layer activations live
// in per-wave bf16 LDS tiles so A fragments are two ds_load_b128 each.

typedef __bf16 bf16_t;
typedef __attribute__((ext_vector_type(16))) __bf16 v16bf;
typedef __attribute__((ext_vector_type(8)))  __bf16 v8bf;
typedef __attribute__((ext_vector_type(8)))  float  v8f;

#define HDIM 128
#define TSTR 136          // bf16 elements per LDS tile row (128 + 8 pad)
#define WCHUNK 16384      // bf16 elements per staged weight chunk (32 KB)

__device__ __forceinline__ void lds_fence() {
  // DS ops are in-order within a wave; this is a compiler fence plus an
  // explicit CDNA5 split-counter wait.
  asm volatile("s_wait_dscnt 0" ::: "memory");
}

// Cooperative 32 KB async copy global->LDS (128 threads x 16 iters x 16 B),
// tracked by ASYNCcnt. Caller barriers afterwards.
__device__ __forceinline__ void stage_32k(const bf16_t* __restrict__ src,
                                          bf16_t* wbuf) {
  const uint64_t ga0 = (uint64_t)(size_t)(src + (size_t)threadIdx.x * 8);
  const uint32_t la0 = (uint32_t)(size_t)(wbuf + (size_t)threadIdx.x * 8);
  #pragma unroll
  for (int it = 0; it < 16; ++it) {
    const uint64_t ga = ga0 + (uint64_t)(it * 2048);
    const uint32_t la = la0 + (uint32_t)(it * 2048);
    asm volatile("global_load_async_to_lds_b128 %0, %1, off"
                 :: "v"(la), "v"(ga) : "memory");
  }
  asm volatile("s_wait_asynccnt 0" ::: "memory");
}

// ---------------------------------------------------------------------------
// Pack an fp32 row-major weight [K x 128] into WMMA B fragments (bf16).
// Tile order: tile = kc*8 + n. Per tile, lane L stores its 16 B-fragment
// values contiguously: col = n*16 + (L&15), k = kc*32 + 8*(L>=16) + i + (i&8).
// ---------------------------------------------------------------------------
__global__ void pack_weights(const float* __restrict__ W, bf16_t* __restrict__ frag) {
  const int tile = blockIdx.x;
  const int kc   = tile >> 3;
  const int n    = tile & 7;
  const int lane = threadIdx.x;            // 0..31
  const int col  = n * 16 + (lane & 15);
  const int khalf = (lane >> 4) * 8;
  bf16_t* dst = frag + ((size_t)tile * 32 + lane) * 16;
  #pragma unroll
  for (int i = 0; i < 16; ++i) {
    const int k = kc * 32 + khalf + i + (i & 8);
    dst[i] = (bf16_t)W[(size_t)k * HDIM + col];
  }
}

// ---------------------------------------------------------------------------
// Device helpers
// ---------------------------------------------------------------------------

// 128-wide K chunk of layer 0: A rows gathered from a global fp32 row,
// B fragments from the staged LDS buffer. Accumulates into acc.
__device__ __forceinline__ void mma_chunk_global(const float* __restrict__ src,
                                                 const bf16_t* __restrict__ wb,
                                                 int lane, v8f* acc) {
  const int hi = lane >> 4;
  #pragma unroll
  for (int kc = 0; kc < 4; ++kc) {
    const int kb = kc * 32 + hi * 8;
    v16bf a;
    #pragma unroll
    for (int i = 0; i < 8; i += 2) {
      const float2 v0 = *(const float2*)(src + kb + i);        // k = kb..kb+7
      const float2 v1 = *(const float2*)(src + kb + 16 + i);   // k = kb+16..kb+23
      a[i]     = (bf16_t)v0.x; a[i + 1] = (bf16_t)v0.y;
      a[8 + i] = (bf16_t)v1.x; a[9 + i] = (bf16_t)v1.y;
    }
    #pragma unroll
    for (int n = 0; n < 8; ++n) {
      const v16bf b = *(const v16bf*)(wb + (((size_t)(kc * 8 + n)) * 32 + lane) * 16);
      acc[n] = __builtin_amdgcn_wmma_f32_16x16x32_bf16(
          false, a, false, b, (short)0, acc[n], false, false);
    }
  }
}

// Hidden layer: x[16x128] (bf16 LDS tile T) @ W[128x128] + b, opt ReLU.
template <bool RELU>
__device__ __forceinline__ void hidden_layer(const bf16_t* __restrict__ T,
                                             const bf16_t* __restrict__ wb,
                                             const float* __restrict__ bias,
                                             int lane, v8f* acc) {
  const int col = lane & 15;
  const int hi  = lane >> 4;
  #pragma unroll
  for (int n = 0; n < 8; ++n) { v8f z = {}; acc[n] = z; }

  #pragma unroll
  for (int kc = 0; kc < 4; ++kc) {
    const int kb = kc * 32 + hi * 8;
    const v8bf lo = *(const v8bf*)(T + col * TSTR + kb);        // k = kb..kb+7
    const v8bf hh = *(const v8bf*)(T + col * TSTR + kb + 16);   // k = kb+16..kb+23
    v16bf a;
    #pragma unroll
    for (int i = 0; i < 8; ++i) { a[i] = lo[i]; a[8 + i] = hh[i]; }
    #pragma unroll
    for (int n = 0; n < 8; ++n) {
      const v16bf b = *(const v16bf*)(wb + (((size_t)(kc * 8 + n)) * 32 + lane) * 16);
      acc[n] = __builtin_amdgcn_wmma_f32_16x16x32_bf16(
          false, a, false, b, (short)0, acc[n], false, false);
    }
  }
  #pragma unroll
  for (int n = 0; n < 8; ++n) {
    const float bn = bias[n * 16 + col];
    #pragma unroll
    for (int j = 0; j < 8; ++j) {
      float v = acc[n][j] + bn;
      acc[n][j] = RELU ? fmaxf(v, 0.f) : v;
    }
  }
}

// Write a C/D-layout tile back to the per-wave bf16 LDS tile (row-major).
__device__ __forceinline__ void store_tile(bf16_t* __restrict__ T,
                                           const v8f* acc, int lane) {
  const int col = lane & 15;
  const int hi  = lane >> 4;
  asm volatile("" ::: "memory");   // keep prior LDS reads before these writes
  #pragma unroll
  for (int n = 0; n < 8; ++n)
    #pragma unroll
    for (int j = 0; j < 8; ++j)
      T[(hi * 8 + j) * TSTR + n * 16 + col] = (bf16_t)acc[n][j];
  lds_fence();
}

// Row-wise LayerNorm over H=128 on a C/D-layout tile (width-16 shfl reduce).
__device__ __forceinline__ void layernorm_tile(v8f* acc,
                                               const float* __restrict__ gamma,
                                               const float* __restrict__ beta,
                                               int lane) {
  const int col = lane & 15;
  float s1[8], s2[8];
  #pragma unroll
  for (int j = 0; j < 8; ++j) { s1[j] = 0.f; s2[j] = 0.f; }
  #pragma unroll
  for (int n = 0; n < 8; ++n)
    #pragma unroll
    for (int j = 0; j < 8; ++j) {
      s1[j] += acc[n][j];
      s2[j] += acc[n][j] * acc[n][j];
    }
  #pragma unroll
  for (int mask = 1; mask < 16; mask <<= 1) {
    #pragma unroll
    for (int j = 0; j < 8; ++j) {
      s1[j] += __shfl_xor(s1[j], mask, 16);
      s2[j] += __shfl_xor(s2[j], mask, 16);
    }
  }
  float mean[8], inv[8];
  #pragma unroll
  for (int j = 0; j < 8; ++j) {
    mean[j] = s1[j] * (1.f / 128.f);
    const float var = s2[j] * (1.f / 128.f) - mean[j] * mean[j];
    inv[j] = rsqrtf(var + 1e-5f);
  }
  #pragma unroll
  for (int n = 0; n < 8; ++n) {
    const float g = gamma[n * 16 + col];
    const float b = beta [n * 16 + col];
    #pragma unroll
    for (int j = 0; j < 8; ++j)
      acc[n][j] = (acc[n][j] - mean[j]) * inv[j] * g + b;
  }
}

// ---------------------------------------------------------------------------
// Edge block: concat[node[s], node[r], edge] (384) -> 4x Linear -> LN.
// ---------------------------------------------------------------------------
__global__ __launch_bounds__(128) void edge_kernel(
    const float* __restrict__ node_attr, const float* __restrict__ edge_attr,
    const int* __restrict__ edge_index,
    const bf16_t* __restrict__ ef0, const bf16_t* __restrict__ ef1,
    const bf16_t* __restrict__ ef2, const bf16_t* __restrict__ ef3,
    const float* __restrict__ b0, const float* __restrict__ b1,
    const float* __restrict__ b2, const float* __restrict__ b3,
    const float* __restrict__ gamma, const float* __restrict__ beta,
    float* __restrict__ agg, float* __restrict__ out_edge, int E) {
  __shared__ __align__(16) bf16_t tiles[4][16 * TSTR];
  __shared__ __align__(32) bf16_t wbuf[WCHUNK];
  const int wave = threadIdx.x >> 5;
  const int lane = threadIdx.x & 31;
  const int tile = blockIdx.x * 4 + wave;
  const int e0   = tile * 16;
  bf16_t* T = tiles[wave];

  const int col = lane & 15;

  // This lane supplies A-row m = col (edge e0+m); clamp for ragged tiles.
  int e = e0 + col;
  if (e >= E) e = E - 1;
  const int s = edge_index[e];
  const int r = edge_index[E + e];
  const float* segs[3] = { node_attr + (size_t)s * HDIM,
                           node_attr + (size_t)r * HDIM,
                           edge_attr + (size_t)e * HDIM };

  v8f acc[8];
  #pragma unroll
  for (int n = 0; n < 8; ++n) { v8f z = {}; acc[n] = z; }

  // ---- layer 0: K = 384 in three 128-wide chunks == concat segments ----
  #pragma unroll
  for (int c = 0; c < 3; ++c) {
    stage_32k(ef0 + (size_t)c * WCHUNK, wbuf);
    __syncthreads();
    mma_chunk_global(segs[c], wbuf, lane, acc);
    __syncthreads();
  }
  #pragma unroll
  for (int n = 0; n < 8; ++n) {
    const float bn = b0[n * 16 + col];
    #pragma unroll
    for (int j = 0; j < 8; ++j)
      acc[n][j] = fmaxf(acc[n][j] + bn, 0.f);
  }
  store_tile(T, acc, lane);

  // ---- hidden layers ----
  stage_32k(ef1, wbuf); __syncthreads();
  hidden_layer<true >(T, wbuf, b1, lane, acc);
  __syncthreads();
  store_tile(T, acc, lane);

  stage_32k(ef2, wbuf); __syncthreads();
  hidden_layer<true >(T, wbuf, b2, lane, acc);
  __syncthreads();
  store_tile(T, acc, lane);

  stage_32k(ef3, wbuf); __syncthreads();
  hidden_layer<false>(T, wbuf, b3, lane, acc);
  layernorm_tile(acc, gamma, beta, lane);
  store_tile(T, acc, lane);

  // ---- epilogue: coalesced residual store + scatter-add (4 cols/lane) ----
  if (e0 < E) {
    const int c0 = lane * 4;
    for (int row = 0; row < 16; ++row) {
      const int er = e0 + row;
      if (er >= E) break;
      const int rr = edge_index[E + er];
      const bf16_t* tp = T + row * TSTR + c0;
      const float u0 = (float)tp[0], u1 = (float)tp[1];
      const float u2 = (float)tp[2], u3 = (float)tp[3];
      const float* ea = edge_attr + (size_t)er * HDIM + c0;
      float*       oe = out_edge + (size_t)er * HDIM + c0;
      float*       ag = agg      + (size_t)rr * HDIM + c0;
      oe[0] = u0 + ea[0]; oe[1] = u1 + ea[1];
      oe[2] = u2 + ea[2]; oe[3] = u3 + ea[3];
      atomicAdd(&ag[0], u0); atomicAdd(&ag[1], u1);
      atomicAdd(&ag[2], u2); atomicAdd(&ag[3], u3);
    }
  }
}

// ---------------------------------------------------------------------------
// Node block: concat[node, agg] (256) -> 4x Linear -> LN; residual output.
// ---------------------------------------------------------------------------
__global__ __launch_bounds__(128) void node_kernel(
    const float* __restrict__ node_attr, const float* __restrict__ agg,
    const bf16_t* __restrict__ nf0, const bf16_t* __restrict__ nf1,
    const bf16_t* __restrict__ nf2, const bf16_t* __restrict__ nf3,
    const float* __restrict__ b0, const float* __restrict__ b1,
    const float* __restrict__ b2, const float* __restrict__ b3,
    const float* __restrict__ gamma, const float* __restrict__ beta,
    float* __restrict__ out_node, int N) {
  __shared__ __align__(16) bf16_t tiles[4][16 * TSTR];
  __shared__ __align__(32) bf16_t wbuf[WCHUNK];
  const int wave = threadIdx.x >> 5;
  const int lane = threadIdx.x & 31;
  const int tile = blockIdx.x * 4 + wave;
  const int n0   = tile * 16;
  bf16_t* T = tiles[wave];

  const int col = lane & 15;

  int nid = n0 + col;
  if (nid >= N) nid = N - 1;
  const float* segs[2] = { node_attr + (size_t)nid * HDIM,
                           agg       + (size_t)nid * HDIM };

  v8f acc[8];
  #pragma unroll
  for (int n = 0; n < 8; ++n) { v8f z = {}; acc[n] = z; }

  // ---- layer 0: K = 256 in two 128-wide chunks == concat segments ----
  #pragma unroll
  for (int c = 0; c < 2; ++c) {
    stage_32k(nf0 + (size_t)c * WCHUNK, wbuf);
    __syncthreads();
    mma_chunk_global(segs[c], wbuf, lane, acc);
    __syncthreads();
  }
  #pragma unroll
  for (int n = 0; n < 8; ++n) {
    const float bn = b0[n * 16 + col];
    #pragma unroll
    for (int j = 0; j < 8; ++j)
      acc[n][j] = fmaxf(acc[n][j] + bn, 0.f);
  }
  store_tile(T, acc, lane);

  stage_32k(nf1, wbuf); __syncthreads();
  hidden_layer<true >(T, wbuf, b1, lane, acc);
  __syncthreads();
  store_tile(T, acc, lane);

  stage_32k(nf2, wbuf); __syncthreads();
  hidden_layer<true >(T, wbuf, b2, lane, acc);
  __syncthreads();
  store_tile(T, acc, lane);

  stage_32k(nf3, wbuf); __syncthreads();
  hidden_layer<false>(T, wbuf, b3, lane, acc);
  layernorm_tile(acc, gamma, beta, lane);
  store_tile(T, acc, lane);

  if (n0 < N) {
    const int c0 = lane * 4;
    for (int row = 0; row < 16; ++row) {
      const int nr = n0 + row;
      if (nr >= N) break;
      const bf16_t* tp = T + row * TSTR + c0;
      const float* na = node_attr + (size_t)nr * HDIM + c0;
      float*       on = out_node + (size_t)nr * HDIM + c0;
      on[0] = (float)tp[0] + na[0]; on[1] = (float)tp[1] + na[1];
      on[2] = (float)tp[2] + na[2]; on[3] = (float)tp[3] + na[3];
    }
  }
}

// ---------------------------------------------------------------------------
extern "C" void kernel_launch(void* const* d_in, const int* in_sizes, int n_in,
                              void* d_out, int out_size, void* d_ws, size_t ws_size,
                              hipStream_t stream) {
  const float* node_attr  = (const float*)d_in[0];
  const float* edge_attr  = (const float*)d_in[1];
  const int*   edge_index = (const int*)d_in[2];
  // d_in[3] = num_nodes (scalar, unused; N derived from in_sizes)
  const float* ew0 = (const float*)d_in[4];  const float* eb0 = (const float*)d_in[5];
  const float* ew1 = (const float*)d_in[6];  const float* eb1 = (const float*)d_in[7];
  const float* ew2 = (const float*)d_in[8];  const float* eb2 = (const float*)d_in[9];
  const float* ew3 = (const float*)d_in[10]; const float* eb3 = (const float*)d_in[11];
  const float* eg  = (const float*)d_in[12]; const float* ebt = (const float*)d_in[13];
  const float* nw0 = (const float*)d_in[14]; const float* nb0 = (const float*)d_in[15];
  const float* nw1 = (const float*)d_in[16]; const float* nb1 = (const float*)d_in[17];
  const float* nw2 = (const float*)d_in[18]; const float* nb2 = (const float*)d_in[19];
  const float* nw3 = (const float*)d_in[20]; const float* nb3 = (const float*)d_in[21];
  const float* ng  = (const float*)d_in[22]; const float* nbt = (const float*)d_in[23];

  const int N = in_sizes[0] / HDIM;
  const int E = in_sizes[1] / HDIM;

  // Workspace: bf16 weight fragments (first ~360 KB), fp32 agg buffer at 1 MB.
  bf16_t* ef0 = (bf16_t*)d_ws;
  bf16_t* ef1 = ef0 + 384 * HDIM;
  bf16_t* ef2 = ef1 + HDIM * HDIM;
  bf16_t* ef3 = ef2 + HDIM * HDIM;
  bf16_t* nf0 = ef3 + HDIM * HDIM;
  bf16_t* nf1 = nf0 + 256 * HDIM;
  bf16_t* nf2 = nf1 + HDIM * HDIM;
  bf16_t* nf3 = nf2 + HDIM * HDIM;
  float*  agg = (float*)((char*)d_ws + (1u << 20));

  pack_weights<<<12 * 8, 32, 0, stream>>>(ew0, ef0);
  pack_weights<<< 4 * 8, 32, 0, stream>>>(ew1, ef1);
  pack_weights<<< 4 * 8, 32, 0, stream>>>(ew2, ef2);
  pack_weights<<< 4 * 8, 32, 0, stream>>>(ew3, ef3);
  pack_weights<<< 8 * 8, 32, 0, stream>>>(nw0, nf0);
  pack_weights<<< 4 * 8, 32, 0, stream>>>(nw1, nf1);
  pack_weights<<< 4 * 8, 32, 0, stream>>>(nw2, nf2);
  pack_weights<<< 4 * 8, 32, 0, stream>>>(nw3, nf3);

  hipMemsetAsync(agg, 0, (size_t)N * HDIM * sizeof(float), stream);

  float* out_node = (float*)d_out;
  float* out_edge = out_node + (size_t)N * HDIM;

  const int etiles = (E + 15) / 16;
  edge_kernel<<<(etiles + 3) / 4, 128, 0, stream>>>(
      node_attr, edge_attr, edge_index, ef0, ef1, ef2, ef3,
      eb0, eb1, eb2, eb3, eg, ebt, agg, out_edge, E);

  const int ntiles = (N + 15) / 16;
  node_kernel<<<(ntiles + 3) / 4, 128, 0, stream>>>(
      node_attr, agg, nf0, nf1, nf2, nf3,
      nb0, nb1, nb2, nb3, ng, nbt, out_node, N);
}
